// SpikeNet_67568425500891
// MI455X (gfx1250) — matrix operand order repaired and checked by reading, not compiled, not run on hardware
//
#include <hip/hip_runtime.h>
#include <hip/hip_bf16.h>

// ---------------------------------------------------------------------------
// SpikeNet forward on gfx1250: bf16 WMMA GEMMs fused with LIF neuron updates.
//   B=256, D=784 (padded 832 = 13*64), T=100, H=4096, O=10 (padded 16)
// Layers 1&2: 32x64 wave tile, double-buffered register pipeline so
// global_load_b128 batches overlap v_wmma_f32_16x16x32_bf16 execution.
// ---------------------------------------------------------------------------

typedef __bf16 bf16;
typedef __attribute__((ext_vector_type(16))) __bf16 v16bf;
typedef __attribute__((ext_vector_type(8)))  __bf16 v8bf;
typedef __attribute__((ext_vector_type(8)))  float  v8f;

union FragB16 { v16bf v; v8bf h[2]; };

// ---------------- utility kernels (once per launch) ----------------

__global__ void zero_f32(float* __restrict__ p, int n) {
  int i = blockIdx.x * blockDim.x + threadIdx.x;
  int stride = gridDim.x * blockDim.x;
  for (; i < n; i += stride) p[i] = 0.0f;
}

// fp32 [srcRows][srcK] -> bf16 [dstRows][dstK], zero-padded
__global__ void cvt_pad_bf16(const float* __restrict__ src, bf16* __restrict__ dst,
                             int srcRows, int srcK, int dstRows, int dstK) {
  long long n = (long long)dstRows * dstK;
  long long i = (long long)blockIdx.x * blockDim.x + threadIdx.x;
  long long stride = (long long)gridDim.x * blockDim.x;
  for (; i < n; i += stride) {
    int row = (int)(i / dstK), col = (int)(i % dstK);
    float v = (row < srcRows && col < srcK) ? src[(long long)row * srcK + col] : 0.0f;
    dst[i] = (bf16)v;
  }
}

// x (B,D,T) fp32 -> xb (T,B,DP) bf16, zero-padded in D.
__global__ void xpose_bf16(const float* __restrict__ x, bf16* __restrict__ xb,
                           int B, int D, int T, int DP) {
  int b = blockIdx.x;
  for (int d = threadIdx.x; d < DP; d += blockDim.x) {
    const float* src = x + ((long long)b * D + d) * T;
    for (int t = 0; t < T; ++t) {
      float v = (d < D) ? src[t] : 0.0f;
      xb[((long long)t * B + b) * DP + d] = (bf16)v;
    }
  }
}

// ---------------- fused GEMM + LIF (layers 1 & 2) ----------------
// C[m][n] = sum_k A[m][k] * W[n][k]   (A: [256][K] bf16, W: [N][K] bf16)
// grid: (N/256, 256/32), block: 128 (4 waves); wave tile = 32 rows x 64 cols
// (2 M-tiles x 4 N-tiles = 8 f32 accumulators).
// K must be a multiple of 64 (pipeline does two 32-deep steps per iteration).
// VGPR striping per CDNA5 ISA 7.12.2:
//   A frag (16x32 bf16): lane half h -> K in {h*8..h*8+7} u {16+h*8..16+h*8+7}
//   B frag (32x16 bf16): lane (r+16h) -> col r, K in {h*8..h*8+7} u {16+h*8..16+h*8+7}
__global__ void __launch_bounds__(128) gemm_lif(
    const bf16* __restrict__ A, const bf16* __restrict__ W,
    const float* __restrict__ bias, float* __restrict__ V,
    bf16* __restrict__ S, int K, int N) {
  const int lane = threadIdx.x & 31;
  const int wave = threadIdx.x >> 5;
  const int hf = lane >> 4;
  const int r  = lane & 15;
  const int m0 = blockIdx.y * 32;
  const int n0 = blockIdx.x * 256 + wave * 64;

  v8f zero8 = {0.f, 0.f, 0.f, 0.f, 0.f, 0.f, 0.f, 0.f};
  v8f acc[2][4];
#pragma unroll
  for (int m = 0; m < 2; ++m)
#pragma unroll
    for (int c = 0; c < 4; ++c) acc[m][c] = zero8;

  const bf16* arow[2];
  arow[0] = A + (long long)(m0 + r) * K;
  arow[1] = A + (long long)(m0 + 16 + r) * K;
  const bf16* wrow[4];
#pragma unroll
  for (int c = 0; c < 4; ++c) wrow[c] = W + (long long)(n0 + c * 16 + r) * K;

  // Double-buffered fragment registers.
  FragB16 uaA[2], ubA[4], uaB[2], ubB[4];

  auto loadf = [&](int k, FragB16 (&ua)[2], FragB16 (&ub)[4]) {
#pragma unroll
    for (int m = 0; m < 2; ++m) {
      ua[m].h[0] = *(const v8bf*)(arow[m] + k + hf * 8);
      ua[m].h[1] = *(const v8bf*)(arow[m] + k + 16 + hf * 8);
    }
#pragma unroll
    for (int c = 0; c < 4; ++c) {
      ub[c].h[0] = *(const v8bf*)(wrow[c] + k + hf * 8);
      ub[c].h[1] = *(const v8bf*)(wrow[c] + k + 16 + hf * 8);
    }
  };
  auto compute = [&](FragB16 (&ua)[2], FragB16 (&ub)[4]) {
#pragma unroll
    for (int m = 0; m < 2; ++m)
#pragma unroll
      for (int c = 0; c < 4; ++c)
        acc[m][c] = __builtin_amdgcn_wmma_f32_16x16x32_bf16(
            false, ua[m].v, false, ub[c].v, (short)0, acc[m][c], false, false);
  };

  // Software pipeline: loads for the next 32-deep step are issued before the
  // WMMAs of the current step, keeping VMEM in flight under the XDL pipe.
  loadf(0, uaA, ubA);
  int k = 0;
  for (; k < K - 64; k += 64) {
    loadf(k + 32, uaB, ubB);
    compute(uaA, ubA);
    loadf(k + 64, uaA, ubA);
    compute(uaB, ubB);
  }
  loadf(k + 32, uaB, ubB);   // tail: k == K-64
  compute(uaA, ubA);
  compute(uaB, ubB);

  // LIF: V += acc + b; spike where V>=1; reset fired to 0.
#pragma unroll
  for (int m = 0; m < 2; ++m) {
#pragma unroll
    for (int c = 0; c < 4; ++c) {
      int col = n0 + c * 16 + r;
      float bv = bias[col];
#pragma unroll
      for (int j = 0; j < 8; ++j) {
        int row = m0 + m * 16 + hf * 8 + j;  // C layout: lane half -> M 0-7/8-15
        long long idx = (long long)row * N + col;
        float v = V[idx] + acc[m][c][j] + bv;
        float sp = (v >= 1.0f) ? 1.0f : 0.0f;
        V[idx] = (v >= 1.0f) ? 0.0f : v;
        S[idx] = (bf16)sp;
      }
    }
  }
}

// ---------------- layer 3: GEMM(K split over waves) + LDS reduce + LIF + count
// W: [16][4096] bf16 (rows 10..15 zero). grid: 16 blocks (M tiles), 128 threads.
__global__ void __launch_bounds__(128) gemm_lif_out(
    const bf16* __restrict__ A, const bf16* __restrict__ W,
    const float* __restrict__ bias, float* __restrict__ V,
    float* __restrict__ sum, int K) {
  const int lane = threadIdx.x & 31;
  const int wave = threadIdx.x >> 5;
  const int hf = lane >> 4;
  const int r  = lane & 15;
  const int m0 = blockIdx.x * 16;
  const int kPer = K >> 2;           // 4 waves split K
  const int k0 = wave * kPer;

  v8f acc = {0.f, 0.f, 0.f, 0.f, 0.f, 0.f, 0.f, 0.f};
  const bf16* arow = A + (long long)(m0 + r) * K;
  const bf16* wrow = W + (long long)r * K;

  for (int k = k0; k < k0 + kPer; k += 32) {
    FragB16 ua, ub;
    ua.h[0] = *(const v8bf*)(arow + k + hf * 8);
    ua.h[1] = *(const v8bf*)(arow + k + 16 + hf * 8);
    ub.h[0] = *(const v8bf*)(wrow + k + hf * 8);
    ub.h[1] = *(const v8bf*)(wrow + k + 16 + hf * 8);
    acc = __builtin_amdgcn_wmma_f32_16x16x32_bf16(
        false, ua.v, false, ub.v, (short)0, acc, false, false);
  }

  __shared__ float red[4][32][8];
#pragma unroll
  for (int j = 0; j < 8; ++j) red[wave][lane][j] = acc[j];
  __syncthreads();

  if (wave == 0) {
    int col = r;
#pragma unroll
    for (int j = 0; j < 8; ++j) {
      float a = red[0][lane][j] + red[1][lane][j] + red[2][lane][j] + red[3][lane][j];
      int row = m0 + hf * 8 + j;
      if (col < 10) {
        int idx = row * 16 + col;
        float v = V[idx] + a + bias[col];
        float sp = (v >= 1.0f) ? 1.0f : 0.0f;
        V[idx] = (v >= 1.0f) ? 0.0f : v;
        sum[idx] += sp;
      }
    }
  }
}

// ---------------- argmax over 10 spike counts (first-max, JAX semantics) ----
__global__ void argmax10(const float* __restrict__ sum, float* __restrict__ out) {
  int b = blockIdx.x * blockDim.x + threadIdx.x;
  if (b < 256) {
    const float* row = sum + b * 16;
    float best = row[0];
    int bi = 0;
    for (int o = 1; o < 10; ++o)
      if (row[o] > best) { best = row[o]; bi = o; }
    out[b] = (float)bi;
  }
}

// ---------------------------------------------------------------------------

extern "C" void kernel_launch(void* const* d_in, const int* in_sizes, int n_in,
                              void* d_out, int out_size, void* d_ws, size_t ws_size,
                              hipStream_t stream) {
  (void)in_sizes; (void)n_in; (void)out_size; (void)ws_size;

  const float* x  = (const float*)d_in[0];
  const float* W1 = (const float*)d_in[1];
  const float* b1 = (const float*)d_in[2];
  const float* W2 = (const float*)d_in[3];
  const float* b2 = (const float*)d_in[4];
  const float* W3 = (const float*)d_in[5];
  const float* b3 = (const float*)d_in[6];
  float* out = (float*)d_out;

  const int B = 256, D = 784, T = 100, H = 4096, O = 10;
  const int DP = 832, OP = 16;   // D padded to 13*64 for the 64-deep pipeline

  // Workspace carve-up (~96 MB), all 256B aligned; V1..sum3 kept contiguous
  // so one zero pass covers all state.
  char* ws = (char*)d_ws;
  size_t off = 0;
  auto carve = [&](size_t bytes) -> void* {
    void* p = ws + off;
    off = (off + bytes + 255) & ~(size_t)255;
    return p;
  };
  float* V1   = (float*)carve((size_t)B * H * 4);
  float* V2   = (float*)carve((size_t)B * H * 4);
  float* V3   = (float*)carve((size_t)B * OP * 4);
  float* sum3 = (float*)carve((size_t)B * OP * 4);
  bf16* s1  = (bf16*)carve((size_t)B * H * 2);
  bf16* s2  = (bf16*)carve((size_t)B * H * 2);
  bf16* W1b = (bf16*)carve((size_t)H * DP * 2);
  bf16* W2b = (bf16*)carve((size_t)H * H * 2);
  bf16* W3b = (bf16*)carve((size_t)OP * H * 2);
  bf16* xb  = (bf16*)carve((size_t)T * B * DP * 2);

  // 1) zero membrane state + spike counters (contiguous region)
  int nZero = 2 * B * H + 2 * B * OP;
  zero_f32<<<1024, 256, 0, stream>>>(V1, nZero);

  // 2) weight conversions (bf16, K/row padded) + input transpose
  cvt_pad_bf16<<<2048, 256, 0, stream>>>(W1, W1b, H, D, H, DP);
  cvt_pad_bf16<<<4096, 256, 0, stream>>>(W2, W2b, H, H, H, H);
  cvt_pad_bf16<<<64,   256, 0, stream>>>(W3, W3b, O, H, OP, H);
  xpose_bf16<<<B, 256, 0, stream>>>(x, xb, B, D, T, DP);

  // 3) T simulation steps, 3 dependent fused GEMM+LIF kernels each
  dim3 blk(128);
  dim3 g12(H / 256, B / 32);   // (16, 8)
  for (int t = 0; t < T; ++t) {
    const bf16* xt = xb + (size_t)t * B * DP;
    gemm_lif<<<g12, blk, 0, stream>>>(xt, W1b, b1, V1, s1, DP, H);
    gemm_lif<<<g12, blk, 0, stream>>>(s1, W2b, b2, V2, s2, H, H);
    gemm_lif_out<<<B / 16, blk, 0, stream>>>(s2, W3b, b3, V3, sum3, H);
  }

  // 4) classification: argmax of accumulated output spikes
  argmax10<<<1, 256, 0, stream>>>(sum3, out);
}
